// BahdanauAttention_47983374631330
// MI455X (gfx1250) — matrix operand order repaired and checked by reading, not compiled, not run on hardware
//
#include <hip/hip_runtime.h>

// ---------------------------------------------------------------------------
// Bahdanau attention for MI455X (gfx1250, wave32, WMMA + TDM).
// t_q=64, t_k=128, b=32, n=1024.
//   PQ = Q @ Wq^T  (2048 x 1024)  -> v_wmma bf16, 16x64 strip per wave
//   PK = K @ Wk^T  (4096 x 1024)  -> v_wmma bf16
//   scores = tanh(PQ+PK+bias).v   -> TDM-staged PK tiles + paired v_tanh_f32
//   softmax (per-wave, registers), context = w @ keys -> v_wmma bf16
// ---------------------------------------------------------------------------

typedef __attribute__((ext_vector_type(16))) __bf16 v16bf;
typedef __attribute__((ext_vector_type(8)))  float  v8f;
typedef __attribute__((ext_vector_type(4)))  unsigned int u32x4;
typedef __attribute__((ext_vector_type(8)))  unsigned int u32x8;

#define TQ 64
#define TK 128
#define BB 32
#define NN 1024
#define QTILE 8     // q rows per block in the score kernel
#define KCHUNK 8    // PK rows per TDM tile

// Two independent CDNA5 TRANS ops back to back: the second tanh is the
// required independent instruction covering the first one's result-use
// hazard, and the consumer of t0 covers the second's. No v_nop needed.
__device__ __forceinline__ void fast_tanh2(float& r0, float& r1,
                                           float x0, float x1) {
  asm volatile("v_tanh_f32 %0, %2\n\tv_tanh_f32 %1, %3"
               : "=v"(r0), "=v"(r1) : "v"(x0), "v"(x1));
}

__device__ __forceinline__ v16bf cvt_frag(const float4& f0, const float4& f1,
                                          const float4& f2, const float4& f3) {
  v16bf A;
  A[0]  = (__bf16)f0.x; A[1]  = (__bf16)f0.y; A[2]  = (__bf16)f0.z; A[3]  = (__bf16)f0.w;
  A[4]  = (__bf16)f1.x; A[5]  = (__bf16)f1.y; A[6]  = (__bf16)f1.z; A[7]  = (__bf16)f1.w;
  A[8]  = (__bf16)f2.x; A[9]  = (__bf16)f2.y; A[10] = (__bf16)f2.z; A[11] = (__bf16)f2.w;
  A[12] = (__bf16)f3.x; A[13] = (__bf16)f3.y; A[14] = (__bf16)f3.z; A[15] = (__bf16)f3.w;
  return A;
}

// A fragment, 16x32 bf16. Lane l: M=l&15, K in {ka..ka+15},
// ka = k0 + (l>=16)*16.
__device__ __forceinline__ v16bf load_a_frag(const float* __restrict__ row, int ka) {
  float4 f0 = *(const float4*)(row + ka);
  float4 f1 = *(const float4*)(row + ka + 4);
  float4 f2 = *(const float4*)(row + ka + 8);
  float4 f3 = *(const float4*)(row + ka + 12);
  return cvt_frag(f0, f1, f2, f3);
}

// Tensor Data Mover: DMA KCHUNK rows of NN f32 (row stride BB*NN elements)
// from global into LDS. 2D descriptor; groups 2/3 unused (zeroed).
__device__ __forceinline__ void tdm_load_rows(const float* __restrict__ gsrc,
                                              unsigned lds_off) {
  unsigned long long ga = (unsigned long long)(uintptr_t)gsrc;
  u32x4 g0;
  g0[0] = 1u;                                     // count=1 (valid user D#)
  g0[1] = lds_off;                                // LDS byte offset
  g0[2] = (unsigned)ga;                           // global_addr[31:0]
  g0[3] = (unsigned)((ga >> 32) & 0x1FFFFFFu) | (2u << 30);  // addr[56:32]|type=2
  u32x8 g1;
  g1[0] = 2u << 16;                               // data_size = 4 bytes
  g1[1] = ((unsigned)NN & 0xFFFFu) << 16;         // tensor_dim0 = 1024 (lo16)
  g1[2] = (((unsigned)NN) >> 16) | ((unsigned)KCHUNK << 16); // dim0 hi | dim1
  g1[3] = ((unsigned)NN) << 16;                   // tile_dim0 = 1024
  g1[4] = (unsigned)KCHUNK;                       // tile_dim1 = 8 rows
  g1[5] = (unsigned)(BB * NN);                    // tensor_dim0_stride (lo32)
  g1[6] = 0u;
  g1[7] = 0u;
  u32x4 gz = {0u, 0u, 0u, 0u};
  asm volatile("tensor_load_to_lds %0, %1, %2, %3"
               :: "s"(g0), "s"(g1), "s"(gz), "s"(gz) : "memory");
}

// v = linear_att / ||linear_att|| * normalize_scalar
__global__ void prep_v_kernel(const float* __restrict__ la,
                              const float* __restrict__ nsc,
                              float* __restrict__ vvec) {
  __shared__ float red[256];
  int t = threadIdx.x;
  float s = 0.f;
  for (int i = t; i < NN; i += 256) { float x = la[i]; s += x * x; }
  red[t] = s;
  __syncthreads();
  for (int off = 128; off > 0; off >>= 1) {
    if (t < off) red[t] += red[t + off];
    __syncthreads();
  }
  float scale = nsc[0] * rsqrtf(red[0]);
  for (int i = t; i < NN; i += 256) vvec[i] = la[i] * scale;
}

// P[M x 1024] = X[M x 1024] * W^T. One wave per 16(M) x 64(N) strip.
// Per K step: issue all 20 b128 loads into distinct raw registers first
// (clause-friendly, progressive loadcnt drain), then convert, then 4 WMMAs.
__global__ void gemm_bf16_wmma(const float* __restrict__ X,
                               const float* __restrict__ W,
                               float* __restrict__ P) {
  const int K = NN, N = NN;
  int lane = threadIdx.x;
  int half = lane >> 4, l15 = lane & 15;
  const float* xrow = X + (size_t)(blockIdx.x * 16 + l15) * K;
  const float* wr   = W + (size_t)(blockIdx.y * 64 + l15) * K;
  v8f a0 = {}, a1 = {}, a2 = {}, a3 = {};
  for (int k0 = 0; k0 < K; k0 += 32) {
    int ka = k0 + half * 16;
    int kb = k0 + half * 16;
    // ---- issue phase: 4 A loads + 16 B loads into distinct registers ----
    float4 ra0 = *(const float4*)(xrow + ka);
    float4 ra1 = *(const float4*)(xrow + ka + 4);
    float4 ra2 = *(const float4*)(xrow + ka + 8);
    float4 ra3 = *(const float4*)(xrow + ka + 12);
    float4 rb[4][4];
#pragma unroll
    for (int nt = 0; nt < 4; ++nt) {
      const float* wrow = wr + (size_t)(16 * nt) * K;
      rb[nt][0] = *(const float4*)(wrow + kb);
      rb[nt][1] = *(const float4*)(wrow + kb + 4);
      rb[nt][2] = *(const float4*)(wrow + kb + 8);
      rb[nt][3] = *(const float4*)(wrow + kb + 12);
    }
    // ---- convert phase ----
    v16bf A = cvt_frag(ra0, ra1, ra2, ra3);
    v16bf Bf[4];
#pragma unroll
    for (int nt = 0; nt < 4; ++nt)
      Bf[nt] = cvt_frag(rb[nt][0], rb[nt][1], rb[nt][2], rb[nt][3]);
    // ---- matrix phase ----
    a0 = __builtin_amdgcn_wmma_f32_16x16x32_bf16(false, A, false, Bf[0], (short)0, a0, false, false);
    a1 = __builtin_amdgcn_wmma_f32_16x16x32_bf16(false, A, false, Bf[1], (short)0, a1, false, false);
    a2 = __builtin_amdgcn_wmma_f32_16x16x32_bf16(false, A, false, Bf[2], (short)0, a2, false, false);
    a3 = __builtin_amdgcn_wmma_f32_16x16x32_bf16(false, A, false, Bf[3], (short)0, a3, false, false);
  }
  float* p = P + (size_t)(blockIdx.x * 16 + half * 8) * N + blockIdx.y * 64 + l15;
#pragma unroll
  for (int r = 0; r < 8; ++r) {
    p[(size_t)r * N]      = a0[r];
    p[(size_t)r * N + 16] = a1[r];
    p[(size_t)r * N + 32] = a2[r];
    p[(size_t)r * N + 48] = a3[r];
  }
}

// One block per (b, 8 q's). pq+bias and v live in registers (32 each/lane),
// PK rows staged into double-buffered LDS by the TDM, paired v_tanh_f32
// inner loop (no hazard NOPs), per-wave register softmax.
__global__ void __launch_bounds__(256)
scores_softmax_tdm(const float* __restrict__ PQ, const float* __restrict__ PK,
                   const float* __restrict__ nbias, const float* __restrict__ vvec,
                   float* __restrict__ Wout) {
  __shared__ float pkb[2][KCHUNK][NN];   // 64 KB
  int b  = blockIdx.x;
  int qt = blockIdx.y;
  int t = threadIdx.x;
  int w = t >> 5, lane = t & 31;

  // Kick off the first TDM tile before touching anything else.
  if (w == 0)
    tdm_load_rows(PK + (size_t)b * NN, (unsigned)(uintptr_t)&pkb[0][0][0]);

  // Register-resident pq row (+bias) and v for this wave's q.
  const int q = qt * QTILE + w;
  const float* pqrow = PQ + (size_t)(q * BB + b) * NN;
  float pq_reg[32], vv_reg[32];
#pragma unroll
  for (int i = 0; i < 32; ++i) {
    int u = lane + 32 * i;
    pq_reg[i] = pqrow[u] + nbias[u];
    vv_reg[i] = vvec[u];
  }

  float x0 = 0.f, x1 = 0.f, x2 = 0.f, x3 = 0.f;  // this lane's 4 scores

  const int NCH = TK / KCHUNK;  // 16 chunks
  for (int c = 0; c < NCH; ++c) {
    if (w == 0) {
      if (c + 1 < NCH) {
        tdm_load_rows(PK + (size_t)((c + 1) * KCHUNK * BB + b) * NN,
                      (unsigned)(uintptr_t)&pkb[(c + 1) & 1][0][0]);
        __builtin_amdgcn_s_wait_tensorcnt(1);   // chunk c has landed
      } else {
        __builtin_amdgcn_s_wait_tensorcnt(0);
      }
    }
    __syncthreads();
    const float* pkc = &pkb[c & 1][0][0];
    for (int j = 0; j < KCHUNK; ++j) {
      const float* pkrow = pkc + (size_t)j * NN;
      float s0 = 0.f, s1 = 0.f;
#pragma unroll
      for (int i = 0; i < 32; i += 2) {
        float t0, t1;
        fast_tanh2(t0, t1,
                   pq_reg[i]     + pkrow[lane + 32 * i],
                   pq_reg[i + 1] + pkrow[lane + 32 * (i + 1)]);
        s0 += t0 * vv_reg[i];
        s1 += t1 * vv_reg[i + 1];
      }
      float s = s0 + s1;
      // butterfly so every lane holds the score, then the owner banks it
      for (int off = 16; off > 0; off >>= 1)
        s += __shfl_xor(s, off, 32);
      int k = c * KCHUNK + j;
      if (lane == (k & 31)) {
        switch (k >> 5) {
          case 0: x0 = s; break;
          case 1: x1 = s; break;
          case 2: x2 = s; break;
          default: x3 = s; break;
        }
      }
    }
    __syncthreads();
  }

  // Per-wave softmax over 128 scores held as 4 regs/lane.
  float m = fmaxf(fmaxf(x0, x1), fmaxf(x2, x3));
  for (int off = 16; off > 0; off >>= 1) m = fmaxf(m, __shfl_xor(m, off, 32));
  x0 = __expf(x0 - m); x1 = __expf(x1 - m);
  x2 = __expf(x2 - m); x3 = __expf(x3 - m);
  float sum = x0 + x1 + x2 + x3;
  for (int off = 16; off > 0; off >>= 1) sum += __shfl_xor(sum, off, 32);
  float inv = 1.f / sum;
  size_t base = ((size_t)q * BB + b) * TK;
  Wout[base + lane]      = x0 * inv;
  Wout[base + 32 + lane] = x1 * inv;
  Wout[base + 64 + lane] = x2 * inv;
  Wout[base + 96 + lane] = x3 * inv;
}

// context[q,b,n] = sum_k w[q,b,k] * keys[k,b,n]; one wave per 16x16 tile,
// K = 128 in four v_wmma bf16 steps.
__global__ void context_wmma(const float* __restrict__ Wt,
                             const float* __restrict__ keys,
                             float* __restrict__ ctx) {
  int qt = blockIdx.x, nt = blockIdx.y, b = blockIdx.z;
  int lane = threadIdx.x, half = lane >> 4, l15 = lane & 15;
  const float* arow = Wt + ((size_t)(qt * 16 + l15) * BB + b) * TK;
  int n = nt * 16 + l15;
  v8f acc = {};
  for (int k0 = 0; k0 < TK; k0 += 32) {
    v16bf A = load_a_frag(arow, k0 + half * 16);
    int kb = k0 + half * 16;
    v16bf B;
#pragma unroll
    for (int j = 0; j < 16; ++j)
      B[j] = (__bf16)keys[((size_t)(kb + j) * BB + b) * NN + n];
    acc = __builtin_amdgcn_wmma_f32_16x16x32_bf16(false, A, false, B,
                                                  (short)0, acc, false, false);
  }
#pragma unroll
  for (int r = 0; r < 8; ++r) {
    int qq = qt * 16 + r + half * 8;
    ctx[((size_t)qq * BB + b) * NN + n] = acc[r];
  }
}

extern "C" void kernel_launch(void* const* d_in, const int* in_sizes, int n_in,
                              void* d_out, int out_size, void* d_ws, size_t ws_size,
                              hipStream_t stream) {
  const float* query = (const float*)d_in[0];  // (64, 32, 1024)
  const float* keys  = (const float*)d_in[1];  // (128, 32, 1024)
  const float* Wq    = (const float*)d_in[2];  // (1024, 1024)
  const float* Wk    = (const float*)d_in[3];  // (1024, 1024)
  const float* la    = (const float*)d_in[4];  // (1024,)
  const float* nsc   = (const float*)d_in[5];  // (1,)
  const float* nbias = (const float*)d_in[6];  // (1024,)

  float* out  = (float*)d_out;
  float* ctx  = out;                          // (t_q, b, n): 64*32*1024
  float* wout = out + (size_t)TQ * BB * NN;   // (t_q, b, t_k): 64*32*128

  float* ws   = (float*)d_ws;
  float* vvec = ws;                           // 1024
  float* PQ   = ws + 1024;                    // 2048 x 1024
  float* PK   = PQ + (size_t)TQ * BB * NN;    // 4096 x 1024

  prep_v_kernel<<<1, 256, 0, stream>>>(la, nsc, vvec);
  gemm_bf16_wmma<<<dim3(TQ * BB / 16, NN / 64), 32, 0, stream>>>(query, Wq, PQ);
  gemm_bf16_wmma<<<dim3(TK * BB / 16, NN / 64), 32, 0, stream>>>(keys, Wk, PK);
  scores_softmax_tdm<<<dim3(BB, TQ / QTILE), 256, 0, stream>>>(PQ, PK, nbias, vvec, wout);
  context_wmma<<<dim3(TQ / 16, NN / 16, BB), 32, 0, stream>>>(wout, keys, ctx);
}
